// TARelationConv_48670569399050
// MI455X (gfx1250) — compile-verified, matrix operand order checked by reading, not run on hardware
//
#include <hip/hip_runtime.h>
#include <hip/hip_bf16.h>

typedef __attribute__((ext_vector_type(16))) _Float16 v16h;
typedef __attribute__((ext_vector_type(8)))  float    v8f;

#define WMMA_F16(a, b, c) \
  __builtin_amdgcn_wmma_f32_16x16x32_f16(false, (a), false, (b), (short)0, (c), false, false)

static __device__ __forceinline__ int laneId() { return (int)(threadIdx.x & 31u); }

// ---- WMMA operand loaders (CDNA5 16x16x32 f16 layouts, wave32) -------------
// A tile (16 x 32) from LDS f16, row stride ldk, K-offset k0.
// lane<16: M=lane, K=k0+{0..7,16..23}; lane>=16: M=lane-16, K=k0+{8..15,24..31}
static __device__ __forceinline__ v16h ldsA(const _Float16* base, int ldk, int k0) {
  const int l = laneId();
  const _Float16* p = base + (l & 15) * ldk + k0 + ((l >> 4) << 3);
  v16h a;
#pragma unroll
  for (int j = 0; j < 8; ++j) { a[j] = p[j]; a[j + 8] = p[j + 16]; }
  return a;
}

// A tile from global f16 row-major (row stride ld)
static __device__ __forceinline__ v16h gAh(const _Float16* X, int ld, int m0, int k0) {
  const int l = laneId();
  const _Float16* p = X + (m0 + (l & 15)) * ld + k0 + ((l >> 4) << 3);
  v16h a;
#pragma unroll
  for (int j = 0; j < 8; ++j) { a[j] = p[j]; a[j + 8] = p[j + 16]; }
  return a;
}

// B tile (32 x 16) from TRANSPOSED f16 weight Wt[N x Kdim] (contiguous in K),
// zero padding k >= Kdim. lane<16: N=n0+lane, K=k0+{0..7,16..23}; etc.
static __device__ __forceinline__ v16h gBt(const _Float16* Wt, int Kdim, int k0, int n0) {
  const int l = laneId();
  const _Float16* p = Wt + (n0 + (l & 15)) * Kdim;
  const int kb = k0 + ((l >> 4) << 3);
  v16h b;
#pragma unroll
  for (int j = 0; j < 8; ++j) {
    const int ka = kb + j, kc = kb + 16 + j;
    b[j]     = (ka < Kdim) ? p[ka] : (_Float16)0.f;
    b[j + 8] = (kc < Kdim) ? p[kc] : (_Float16)0.f;
  }
  return b;
}

// ---- Kernel 0: one-time f32 -> f16 conversion (weights transposed) ---------
struct Seg { const float* src; _Float16* dst; int rows, cols, tr; };
struct CvtArgs { Seg s[8]; };

__global__ __launch_bounds__(256) void cvt_kernel(CvtArgs a) {
  const int tid = blockIdx.x * 256 + threadIdx.x;
  const int stride = gridDim.x * 256;
  for (int g = 0; g < 8; ++g) {
    const Seg sg = a.s[g];
    const int n = sg.rows * sg.cols;
    if (sg.tr) {
      for (int i = tid; i < n; i += stride) {
        const int r = i / sg.cols, c = i - r * sg.cols;
        sg.dst[c * sg.rows + r] = (_Float16)sg.src[i];
      }
    } else {
      for (int i = tid; i < n; i += stride) sg.dst[i] = (_Float16)sg.src[i];
    }
  }
}

// ---- Kernel 1: batch-aware kNN (K=16, ascending d2, tie -> low idx) --------
__global__ __launch_bounds__(256) void knn_kernel(const float* __restrict__ xyz,
                                                  int* __restrict__ colw) {
  const int q = blockIdx.x;
  const int base = (q >> 10) << 10;
  __shared__ float d2s[1024];
  __shared__ float sval[8];
  __shared__ int   sidx[8];
  const int t = threadIdx.x, w = t >> 5, l = t & 31;
  const float qx = xyz[q * 3 + 0], qy = xyz[q * 3 + 1], qz = xyz[q * 3 + 2];
  for (int c = t; c < 1024; c += 256) {
    const float dx = qx - xyz[(base + c) * 3 + 0];
    const float dy = qy - xyz[(base + c) * 3 + 1];
    const float dz = qz - xyz[(base + c) * 3 + 2];
    d2s[c] = dx * dx + dy * dy + dz * dz;
  }
  __syncthreads();
  for (int it = 0; it < 16; ++it) {
    float bv = 3.4e38f; int bi = 0x7fffffff;
    for (int c = t; c < 1024; c += 256) {
      const float v = d2s[c];
      if (v < bv || (v == bv && c < bi)) { bv = v; bi = c; }
    }
#pragma unroll
    for (int s = 16; s > 0; s >>= 1) {          // wave32 min-reduce
      const float ov = __shfl_xor(bv, s, 32);
      const int   oi = __shfl_xor(bi, s, 32);
      if (ov < bv || (ov == bv && oi < bi)) { bv = ov; bi = oi; }
    }
    if (l == 0) { sval[w] = bv; sidx[w] = bi; }
    __syncthreads();
    if (t == 0) {
      float fv = sval[0]; int fi = sidx[0];
#pragma unroll
      for (int j = 1; j < 8; ++j)
        if (sval[j] < fv || (sval[j] == fv && sidx[j] < fi)) { fv = sval[j]; fi = sidx[j]; }
      colw[q * 16 + it] = base + fi;
      d2s[fi] = 3.4e38f;
    }
    __syncthreads();
  }
}

// ---- Kernel 2: lang encoder (Linear -> BatchNorm -> ReLU -> Linear) --------
__global__ __launch_bounds__(256) void lang_kernel(
    const _Float16* __restrict__ Xlh,                   // [64,768] f16
    const float* __restrict__ bl1,
    const float* __restrict__ bn_g, const float* __restrict__ bn_b,
    const float* __restrict__ bl2,
    const _Float16* __restrict__ Wl1t,                  // [128,768] f16 (N-major)
    const _Float16* __restrict__ Wl2t,                  // [128,128] f16 (N-major)
    float* __restrict__ lang,                           // [64,128] f32
    _Float16* __restrict__ langh,                       // [64,128] f16 row-major
    _Float16* __restrict__ langhT)                      // [4][128][16] f16 feat-major
{
  __shared__ float    hl[64 * 128];
  __shared__ _Float16 hh[64 * 128];
  __shared__ float mu[128], rs[128];
  const int t = threadIdx.x, w = t >> 5, l = t & 31;
  const int n0 = 16 * w;
  v8f a0 = {}, a1 = {}, a2 = {}, a3 = {};
  for (int kk = 0; kk < 24; ++kk) {
    const v16h b = gBt(Wl1t, 768, 32 * kk, n0);
    a0 = WMMA_F16(gAh(Xlh, 768,  0, 32 * kk), b, a0);
    a1 = WMMA_F16(gAh(Xlh, 768, 16, 32 * kk), b, a1);
    a2 = WMMA_F16(gAh(Xlh, 768, 32, 32 * kk), b, a2);
    a3 = WMMA_F16(gAh(Xlh, 768, 48, 32 * kk), b, a3);
  }
  {
    const int nn = n0 + (l & 15), mb = (l >> 4) << 3;
    const float bias = bl1[nn];
#pragma unroll
    for (int v = 0; v < 8; ++v) {
      hl[(mb + v)      * 128 + nn] = a0[v] + bias;
      hl[(mb + v + 16) * 128 + nn] = a1[v] + bias;
      hl[(mb + v + 32) * 128 + nn] = a2[v] + bias;
      hl[(mb + v + 48) * 128 + nn] = a3[v] + bias;
    }
  }
  __syncthreads();
  if (t < 128) {                      // batch stats over 64 rows (biased var)
    float s = 0.f, s2 = 0.f;
    for (int r = 0; r < 64; ++r) { const float v = hl[r * 128 + t]; s += v; s2 += v * v; }
    const float m = s * (1.f / 64.f);
    mu[t] = m; rs[t] = rsqrtf(s2 * (1.f / 64.f) - m * m + 1e-5f);
  }
  __syncthreads();
  for (int i = t; i < 64 * 128; i += 256) {
    const int c = i & 127;
    const float v = (hl[i] - mu[c]) * rs[c] * bn_g[c] + bn_b[c];
    hh[i] = (_Float16)fmaxf(v, 0.f);
  }
  __syncthreads();
  v8f o0 = {}, o1 = {}, o2 = {}, o3 = {};
  for (int kk = 0; kk < 4; ++kk) {
    const v16h b = gBt(Wl2t, 128, 32 * kk, n0);
    o0 = WMMA_F16(ldsA(hh +  0 * 128, 128, 32 * kk), b, o0);
    o1 = WMMA_F16(ldsA(hh + 16 * 128, 128, 32 * kk), b, o1);
    o2 = WMMA_F16(ldsA(hh + 32 * 128, 128, 32 * kk), b, o2);
    o3 = WMMA_F16(ldsA(hh + 48 * 128, 128, 32 * kk), b, o3);
  }
  {
    const int nn = n0 + (l & 15), mb = (l >> 4) << 3;
    const float bias = bl2[nn];
#pragma unroll
    for (int v = 0; v < 8; ++v) {
      const float r0v = o0[v] + bias, r1v = o1[v] + bias;
      const float r2v = o2[v] + bias, r3v = o3[v] + bias;
      const int m0r = mb + v, m1r = mb + v + 16, m2r = mb + v + 32, m3r = mb + v + 48;
      lang[m0r * 128 + nn] = r0v; lang[m1r * 128 + nn] = r1v;
      lang[m2r * 128 + nn] = r2v; lang[m3r * 128 + nn] = r3v;
      langh[m0r * 128 + nn] = (_Float16)r0v; langh[m1r * 128 + nn] = (_Float16)r1v;
      langh[m2r * 128 + nn] = (_Float16)r2v; langh[m3r * 128 + nn] = (_Float16)r3v;
      langhT[(m0r >> 4) * 2048 + nn * 16 + (m0r & 15)] = (_Float16)r0v;
      langhT[(m1r >> 4) * 2048 + nn * 16 + (m1r & 15)] = (_Float16)r1v;
      langhT[(m2r >> 4) * 2048 + nn * 16 + (m2r & 15)] = (_Float16)r2v;
      langhT[(m3r >> 4) * 2048 + nn * 16 + (m3r & 15)] = (_Float16)r3v;
    }
  }
}

// ---- Kernel 3: feat encoder (Linear->LN->ReLU->Linear) + attention logits --
__global__ __launch_bounds__(256) void feat_kernel(
    const _Float16* __restrict__ Xh16,                  // [4096,256] f16
    const float* __restrict__ b1,
    const float* __restrict__ ln_g, const float* __restrict__ ln_b,
    const float* __restrict__ b2,
    const _Float16* __restrict__ W1t,                   // [128,256] f16 N-major
    const _Float16* __restrict__ W2t,                   // [128,128] f16 N-major
    const _Float16* __restrict__ langh,                 // [64,128] f16 row-major
    float* __restrict__ feats,                          // [4096,128]
    float* __restrict__ atten)                          // [4096,16]
{
  __shared__ __align__(16) _Float16 Xh[16 * 256];
  __shared__ float    h[16 * 128];
  __shared__ _Float16 hh[16 * 128];
  __shared__ _Float16 fh[16 * 128];
  __shared__ float mu[16], rs[16];
  const int t = threadIdx.x, w = t >> 5, l = t & 31;
  const int r0 = blockIdx.x * 16;
  {                                   // stage A tile: b128 copies, no convert
    const uint4* src = (const uint4*)(Xh16 + r0 * 256);
    uint4* dst = (uint4*)Xh;
    for (int i = t; i < 16 * 256 / 8; i += 256) dst[i] = src[i];
  }
  __syncthreads();
  const int n0 = 16 * w;
  v8f acc = {};
  for (int kk = 0; kk < 8; ++kk) {
    if (kk + 1 < 8)
      __builtin_prefetch(W1t + (n0 + (l & 15)) * 256 + 32 * (kk + 1), 0, 1);
    acc = WMMA_F16(ldsA(Xh, 256, 32 * kk), gBt(W1t, 256, 32 * kk, n0), acc);
  }
  {
    const int nn = n0 + (l & 15), mb = (l >> 4) << 3;
    const float bias = b1[nn];
#pragma unroll
    for (int v = 0; v < 8; ++v) h[(mb + v) * 128 + nn] = acc[v] + bias;
  }
  __syncthreads();
  if (t < 16) {                       // LayerNorm stats per row
    float s = 0.f, s2 = 0.f;
    for (int c = 0; c < 128; ++c) { const float v = h[t * 128 + c]; s += v; s2 += v * v; }
    const float m = s * (1.f / 128.f);
    mu[t] = m; rs[t] = rsqrtf(s2 * (1.f / 128.f) - m * m + 1e-5f);
  }
  __syncthreads();
  for (int i = t; i < 16 * 128; i += 256) {
    const int r = i >> 7, c = i & 127;
    const float v = (h[i] - mu[r]) * rs[r] * ln_g[c] + ln_b[c];
    hh[i] = (_Float16)fmaxf(v, 0.f);
  }
  __syncthreads();
  v8f o = {};
  for (int kk = 0; kk < 4; ++kk)
    o = WMMA_F16(ldsA(hh, 128, 32 * kk), gBt(W2t, 128, 32 * kk, n0), o);
  {
    const int nn = n0 + (l & 15), mb = (l >> 4) << 3;
    const float bias = b2[nn];
#pragma unroll
    for (int v = 0; v < 8; ++v) {
      const float fv = o[v] + bias;
      feats[(r0 + mb + v) * 128 + nn] = fv;
      fh[(mb + v) * 128 + nn] = (_Float16)fv;
    }
  }
  __syncthreads();
  if (w == 0) {                       // atten tile = feats[16,128] @ lang_b^T
    const _Float16* lbh = langh + (r0 >> 10) * 16 * 128;
    v8f at = {};
    for (int kk = 0; kk < 4; ++kk) {
      const int nn = l & 15;
      const int kb = 32 * kk + ((l >> 4) << 3);
      const _Float16* p = lbh + nn * 128 + kb;   // B[k=feat][n=tok], contiguous in k
      v16h b;
#pragma unroll
      for (int j = 0; j < 8; ++j) { b[j] = p[j]; b[j + 8] = p[j + 16]; }
      at = WMMA_F16(ldsA(fh, 128, 32 * kk), b, at);
    }
    const int tok = l & 15, mb = (l >> 4) << 3;
#pragma unroll
    for (int v = 0; v < 8; ++v) atten[(r0 + mb + v) * 16 + tok] = at[v];
  }
}

// ---- Kernel 4: per-query edges: softmax, ctx/ew WMMA GEMMs, message sum ----
__global__ __launch_bounds__(256) void edge_kernel(
    const float* __restrict__ xyz,
    const float* __restrict__ mask,                     // [4096,16]
    const float* __restrict__ feats,                    // [4096,128]
    const float* __restrict__ atten,                    // [4096,16]
    const _Float16* __restrict__ langhT,                // [4][128][16]
    const int*   __restrict__ colw,                     // [4096,16]
    const float* __restrict__ br1, const float* __restrict__ br2,
    const _Float16* __restrict__ Wr1t,                  // [64,10] f16 N-major
    const _Float16* __restrict__ Wr2t,                  // [128,64] f16 N-major
    float* __restrict__ out)                            // [4096,128]
{
  __shared__ int      cols[16];
  __shared__ float    attf[16 * 16];
  __shared__ _Float16 attA[16 * 32];
  __shared__ _Float16 geomA[16 * 32];
  __shared__ _Float16 ewh[16 * 64];
  __shared__ float    fnb[16 * 128];
  __shared__ float    maxl[16], suml[16];
  const int t = threadIdx.x, w = t >> 5, l = t & 31;
  const int q = blockIdx.x;
  if (t < 16) cols[t] = colw[q * 16 + t];
  __syncthreads();
  {                                    // gather neighbor logits
    const int k = t >> 4, tok = t & 15;
    attf[t] = atten[cols[k] * 16 + tok];
  }
  for (int i = t; i < 16 * 128; i += 256) {   // gather neighbor features
    const int k = i >> 7, c = i & 127;
    fnb[i] = feats[cols[k] * 128 + c];
  }
  if (t < 16) {                        // edge geometry row [10], pad K to 32
    const float xi0 = xyz[q * 3], xi1 = xyz[q * 3 + 1], xi2 = xyz[q * 3 + 2];
    const int cj = cols[t];
    const float xj0 = xyz[cj * 3], xj1 = xyz[cj * 3 + 1], xj2 = xyz[cj * 3 + 2];
    const float d0 = xi0 - xj0, d1 = xi1 - xj1, dd2 = xi2 - xj2;
    const float nr = sqrtf(d0 * d0 + d1 * d1 + dd2 * dd2 + 1e-12f);
    const float g[10] = {xi0, xi1, xi2, xj0, xj1, xj2, d0, d1, dd2, nr};
#pragma unroll
    for (int j = 0; j < 10; ++j) geomA[t * 32 + j] = (_Float16)g[j];
#pragma unroll
    for (int j = 10; j < 32; ++j) geomA[t * 32 + j] = (_Float16)0.f;
  }
  __syncthreads();
  if (t < 16) {                        // per-token max over 16 edges
    float mx = -3.4e38f;
    for (int k = 0; k < 16; ++k) mx = fmaxf(mx, attf[k * 16 + t]);
    maxl[t] = mx;
  }
  __syncthreads();
  attf[t] = __expf(attf[t] - maxl[t & 15]);
  __syncthreads();
  if (t < 16) {
    float s = 0.f;
    for (int k = 0; k < 16; ++k) s += attf[k * 16 + t];
    suml[t] = s;
  }
  __syncthreads();
  attf[t] = attf[t] / suml[t & 15] * mask[q * 16 + (t & 15)];
  __syncthreads();
  if (t < 16) {                        // renormalize per edge over tokens
    float s = 0.f;
    for (int tok = 0; tok < 16; ++tok) s += attf[t * 16 + tok];
    const float rcp = 1.f / (s + 1e-7f);
#pragma unroll
    for (int tok = 0; tok < 16; ++tok) attA[t * 32 + tok] = (_Float16)(attf[t * 16 + tok] * rcp);
#pragma unroll
    for (int tok = 16; tok < 32; ++tok) attA[t * 32 + tok] = (_Float16)0.f;
  }
  __syncthreads();
  const _Float16* lbT = langhT + (q >> 10) * 2048;
  const int n0 = 16 * w;
  v8f z = {};
  const v8f ctx = WMMA_F16(ldsA(attA, 32, 0), gBt(lbT, 16, 0, n0), z);  // K=16 pad
  if (w < 4) {                         // rel_encoder GEMM1 (N=64, K=10 pad)
    const v8f e1 = WMMA_F16(ldsA(geomA, 32, 0), gBt(Wr1t, 10, 0, n0), z);
    const int nn = n0 + (l & 15), mb = (l >> 4) << 3;
    const float bias = br1[nn];
#pragma unroll
    for (int v = 0; v < 8; ++v) ewh[(mb + v) * 64 + nn] = (_Float16)fmaxf(e1[v] + bias, 0.f);
  }
  __syncthreads();
  v8f e2 = {};
  for (int kk = 0; kk < 2; ++kk)       // rel_encoder GEMM2 (K=64)
    e2 = WMMA_F16(ldsA(ewh, 64, 32 * kk), gBt(Wr2t, 64, 32 * kk, n0), e2);
  {                                    // msg = feats[col]*ctx*ew ; sum 16 edges
    const int nn = n0 + (l & 15), mb = (l >> 4) << 3;
    const float b2v = br2[nn];
    float part = 0.f;
#pragma unroll
    for (int v = 0; v < 8; ++v)
      part += fnb[(mb + v) * 128 + nn] * ctx[v] * (e2[v] + b2v);
    part += __shfl_xor(part, 16, 32);  // combine M=0..7 and M=8..15 halves
    if (l < 16) out[q * 128 + nn] = feats[q * 128 + nn] + part;
  }
}

extern "C" void kernel_launch(void* const* d_in, const int* in_sizes, int n_in,
                              void* d_out, int out_size, void* d_ws, size_t ws_size,
                              hipStream_t stream) {
  (void)in_sizes; (void)n_in; (void)out_size; (void)ws_size;
  const float* xyz  = (const float*)d_in[0];
  // d_in[1] = batch_index (implied by q/1024, unused)
  const float* X    = (const float*)d_in[2];
  const float* Xl   = (const float*)d_in[3];
  const float* mask = (const float*)d_in[4];
  const float* W1   = (const float*)d_in[5];  const float* b1   = (const float*)d_in[6];
  const float* ln_g = (const float*)d_in[7];  const float* ln_b = (const float*)d_in[8];
  const float* W2   = (const float*)d_in[9];  const float* b2   = (const float*)d_in[10];
  const float* Wl1  = (const float*)d_in[11]; const float* bl1  = (const float*)d_in[12];
  const float* bn_g = (const float*)d_in[13]; const float* bn_b = (const float*)d_in[14];
  const float* Wl2  = (const float*)d_in[15]; const float* bl2  = (const float*)d_in[16];
  const float* Wr1  = (const float*)d_in[17]; const float* br1  = (const float*)d_in[18];
  const float* Wr2  = (const float*)d_in[19]; const float* br2  = (const float*)d_in[20];

  char* ws = (char*)d_ws;               // all offsets 256-aligned
  int*      colw   = (int*)      (ws + 0);        // 4096*16 i32   (262144 B)
  float*    lang   = (float*)    (ws + 262144);   // 64*128 f32    (32768 B)
  float*    feats  = (float*)    (ws + 294912);   // 4096*128 f32  (2097152 B)
  float*    atten  = (float*)    (ws + 2392064);  // 4096*16 f32   (262144 B)
  _Float16* langh  = (_Float16*) (ws + 2654208);  // 64*128 f16    (16384 B)
  _Float16* Xh16   = (_Float16*) (ws + 2670592);  // 4096*256 f16  (2097152 B)
  _Float16* Xlh    = (_Float16*) (ws + 4767744);  // 64*768 f16    (98304 B)
  _Float16* W1t    = (_Float16*) (ws + 4866048);  // 128*256 f16   (65536 B)
  _Float16* W2t    = (_Float16*) (ws + 4931584);  // 128*128 f16   (32768 B)
  _Float16* Wl1t   = (_Float16*) (ws + 4964352);  // 128*768 f16   (196608 B)
  _Float16* Wl2t   = (_Float16*) (ws + 5160960);  // 128*128 f16   (32768 B)
  _Float16* Wr1t   = (_Float16*) (ws + 5193728);  // 64*10 f16     (pad 2048 B)
  _Float16* Wr2t   = (_Float16*) (ws + 5195776);  // 128*64 f16    (16384 B)
  _Float16* langhT = (_Float16*) (ws + 5212160);  // 4*128*16 f16  (16384 B)
  float* out = (float*)d_out;

  CvtArgs ca;
  ca.s[0] = {X,   Xh16, 4096, 256, 0};
  ca.s[1] = {Xl,  Xlh,    64, 768, 0};
  ca.s[2] = {W1,  W1t,   256, 128, 1};
  ca.s[3] = {W2,  W2t,   128, 128, 1};
  ca.s[4] = {Wl1, Wl1t,  768, 128, 1};
  ca.s[5] = {Wl2, Wl2t,  128, 128, 1};
  ca.s[6] = {Wr1, Wr1t,   10,  64, 1};
  ca.s[7] = {Wr2, Wr2t,   64, 128, 1};

  cvt_kernel <<<1024, 256, 0, stream>>>(ca);
  knn_kernel <<<4096, 256, 0, stream>>>(xyz, colw);
  lang_kernel<<<1,    256, 0, stream>>>(Xlh, bl1, bn_g, bn_b, bl2, Wl1t, Wl2t,
                                        lang, langh, langhT);
  feat_kernel<<<256,  256, 0, stream>>>(Xh16, b1, ln_g, ln_b, b2, W1t, W2t, langh,
                                        feats, atten);
  edge_kernel<<<4096, 256, 0, stream>>>(xyz, mask, feats, atten, langhT, colw,
                                        br1, br2, Wr1t, Wr2t, out);
}
